// QTAttB_55602646614239
// MI455X (gfx1250) — matrix operand order repaired and checked by reading, not compiled, not run on hardware
//
#include <hip/hip_runtime.h>

#define N_TOK   2048
#define IN_DIM  256
#define DIM     256
#define NHEADS  8
#define CH      32
#define QKV_DIM 768
#define TOPK    16

typedef __attribute__((ext_vector_type(16))) _Float16 v16h;
typedef __attribute__((ext_vector_type(8)))  _Float16 v8h;
typedef __attribute__((ext_vector_type(8)))  float    v8f;

// Load 16 f16 as two 16-byte chunks (A- and B-operand both use chunks 16 elems apart
// -> pass p1 = p0 + 16).
__device__ __forceinline__ v16h ld_pair(const _Float16* p0, const _Float16* p1) {
  v8h lo = *(const v8h*)p0;
  v8h hi = *(const v8h*)p1;
  v16h r;
#pragma unroll
  for (int i = 0; i < 8; ++i) { r[i] = lo[i]; r[i + 8] = hi[i]; }
  return r;
}

#define WMMA_F16(a, b, c) \
  __builtin_amdgcn_wmma_f32_16x16x32_f16(false, (a), false, (b), (short)0, (c), false, false)

// ---------- LayerNorm 1: fp32 in -> f16 out ----------
__global__ __launch_bounds__(256) void ln1_kernel(const float* __restrict__ x,
                                                  const float* __restrict__ w,
                                                  const float* __restrict__ b,
                                                  _Float16* __restrict__ out) {
  __shared__ float red[256];
  int row = blockIdx.x, tid = threadIdx.x;
  float v = x[row * IN_DIM + tid];
  red[tid] = v; __syncthreads();
  for (int s = 128; s; s >>= 1) { if (tid < s) red[tid] += red[tid + s]; __syncthreads(); }
  float mu = red[0] * (1.0f / IN_DIM); __syncthreads();
  float d = v - mu;
  red[tid] = d * d; __syncthreads();
  for (int s = 128; s; s >>= 1) { if (tid < s) red[tid] += red[tid + s]; __syncthreads(); }
  float var = red[0] * (1.0f / IN_DIM);
  float y = d * rsqrtf(var + 1e-5f) * w[tid] + b[tid];
  out[row * IN_DIM + tid] = (_Float16)y;
}

// ---------- LayerNorm 2: fp32 in -> fp32 out ----------
__global__ __launch_bounds__(256) void ln2_kernel(const float* __restrict__ x,
                                                  const float* __restrict__ w,
                                                  const float* __restrict__ b,
                                                  float* __restrict__ out) {
  __shared__ float red[256];
  int row = blockIdx.x, tid = threadIdx.x;
  float v = x[row * DIM + tid];
  red[tid] = v; __syncthreads();
  for (int s = 128; s; s >>= 1) { if (tid < s) red[tid] += red[tid + s]; __syncthreads(); }
  float mu = red[0] * (1.0f / DIM); __syncthreads();
  float d = v - mu;
  red[tid] = d * d; __syncthreads();
  for (int s = 128; s; s >>= 1) { if (tid < s) red[tid] += red[tid + s]; __syncthreads(); }
  float var = red[0] * (1.0f / DIM);
  out[row * DIM + tid] = d * rsqrtf(var + 1e-5f) * w[tid] + b[tid];
}

// ---------- Transpose + convert weights to f16 ([n][k] layout for B-operand) ----------
__global__ void convw_kernel(const float* __restrict__ wqkv, const float* __restrict__ wproj,
                             _Float16* __restrict__ wqkvT, _Float16* __restrict__ wprojT) {
  int i = blockIdx.x * blockDim.x + threadIdx.x;
  if (i < QKV_DIM * IN_DIM) {
    int n = i / IN_DIM, k = i % IN_DIM;
    wqkvT[i] = (_Float16)wqkv[k * QKV_DIM + n];
  }
  if (i < DIM * DIM) {
    int n = i / DIM, k = i % DIM;
    wprojT[i] = (_Float16)wproj[k * DIM + n];
  }
}

// ---------- QKV GEMM: (2048x256) @ (256x768), split into q16/k16/v32+vT16 ----------
__global__ __launch_bounds__(32) void qkv_gemm_kernel(const _Float16* __restrict__ x16,
                                                      const _Float16* __restrict__ wT,
                                                      _Float16* __restrict__ q16,
                                                      _Float16* __restrict__ k16,
                                                      _Float16* __restrict__ vT16,
                                                      float* __restrict__ v32) {
  int tm = blockIdx.x, tn = blockIdx.y;
  int lane = threadIdx.x;
  int rowA = tm * 16 + (lane & 15);
  int aoff = (lane >> 4) * 8;
  int colB = tn * 16 + (lane & 15);
  int boff = (lane >> 4) * 8;
  const _Float16* ap = x16 + rowA * IN_DIM;
  const _Float16* bp = wT + colB * IN_DIM;
  v8f acc = {};
#pragma unroll
  for (int kb = 0; kb < IN_DIM; kb += 32) {
    v16h a = ld_pair(ap + kb + aoff, ap + kb + aoff + 16);
    v16h b = ld_pair(bp + kb + boff, bp + kb + boff + 16);
    acc = WMMA_F16(a, b, acc);
  }
  int col = colB;                     // 0..767 : [q | k | v]
  int rbase = tm * 16 + (lane >> 4) * 8;
#pragma unroll
  for (int j = 0; j < 8; ++j) {
    int row = rbase + j;
    float val = acc[j];
    if (col < 256) {
      q16[row * DIM + col] = (_Float16)val;
    } else if (col < 512) {
      k16[row * DIM + (col - 256)] = (_Float16)val;
    } else {
      int d = col - 512;
      v32[row * DIM + d] = val;
      vT16[(size_t)d * N_TOK + row] = (_Float16)val;
    }
  }
}

// ---------- Per-head scores S = Q_h @ K_h^T (16x64 strip per wave) ----------
__global__ __launch_bounds__(32) void qk_kernel(const _Float16* __restrict__ q16,
                                                const _Float16* __restrict__ k16,
                                                float* __restrict__ S, int h) {
  int tm = blockIdx.x, tg = blockIdx.y;
  int lane = threadIdx.x;
  int rowA = tm * 16 + (lane & 15);
  const _Float16* ap = q16 + rowA * DIM + h * CH + (lane >> 4) * 8;
  v16h a = ld_pair(ap, ap + 16);
  int rbase = tm * 16 + (lane >> 4) * 8;
#pragma unroll
  for (int t = 0; t < 4; ++t) {
    int colS = tg * 64 + t * 16 + (lane & 15);
    const _Float16* bp = k16 + colS * DIM + h * CH + (lane >> 4) * 8;
    v16h b = ld_pair(bp, bp + 16);
    v8f acc = {};
    acc = WMMA_F16(a, b, acc);
#pragma unroll
    for (int j = 0; j < 8; ++j)
      S[(size_t)(rbase + j) * N_TOK + colS] = acc[j];
  }
}

// ---------- Row softmax(S*Ch^-0.5), write P16, accumulate A_mean += P/8 ----------
__global__ __launch_bounds__(256) void softmax_kernel(const float* __restrict__ S,
                                                      _Float16* __restrict__ P16,
                                                      float* __restrict__ Amean,
                                                      int init) {
  __shared__ float red[256];
  int row = blockIdx.x, tid = threadIdx.x;
  const float* srow = S + (size_t)row * N_TOK;
  const float scale = 0.17677669529663687f;  // 1/sqrt(32)
  float v[8];
  float m = -3.4e38f;
#pragma unroll
  for (int i = 0; i < 8; ++i) { v[i] = srow[i * 256 + tid]; m = fmaxf(m, v[i]); }
  red[tid] = m; __syncthreads();
  for (int s = 128; s; s >>= 1) { if (tid < s) red[tid] = fmaxf(red[tid], red[tid + s]); __syncthreads(); }
  float vmax = red[0]; __syncthreads();
  float sum = 0.f;
#pragma unroll
  for (int i = 0; i < 8; ++i) { v[i] = __expf(scale * (v[i] - vmax)); sum += v[i]; }
  red[tid] = sum; __syncthreads();
  for (int s = 128; s; s >>= 1) { if (tid < s) red[tid] += red[tid + s]; __syncthreads(); }
  float inv = 1.f / red[0];
#pragma unroll
  for (int i = 0; i < 8; ++i) {
    size_t idx = (size_t)row * N_TOK + i * 256 + tid;
    float p = v[i] * inv;
    P16[idx] = (_Float16)p;
    float c = p * 0.125f;
    if (init) Amean[idx] = c; else Amean[idx] += c;
  }
}

// ---------- Per-head AV: msg[:, h*32:(h+1)*32] = P @ V_h  (16x32 tile per wave) ----------
__global__ __launch_bounds__(32) void av_kernel(const _Float16* __restrict__ P16,
                                                const _Float16* __restrict__ vT16,
                                                _Float16* __restrict__ msg16, int h) {
  int tm = blockIdx.x;
  int lane = threadIdx.x;
  int rowA = tm * 16 + (lane & 15);
  const _Float16* ap = P16 + (size_t)rowA * N_TOK + (lane >> 4) * 8;
  const _Float16* bp0 = vT16 + (size_t)(h * CH + (lane & 15)) * N_TOK + (lane >> 4) * 8;
  const _Float16* bp1 = bp0 + (size_t)16 * N_TOK;
  v8f acc0 = {}, acc1 = {};
#pragma unroll 4
  for (int sb = 0; sb < N_TOK; sb += 32) {
    v16h a  = ld_pair(ap + sb, ap + sb + 16);
    v16h b0 = ld_pair(bp0 + sb, bp0 + sb + 16);
    v16h b1 = ld_pair(bp1 + sb, bp1 + sb + 16);
    acc0 = WMMA_F16(a, b0, acc0);
    acc1 = WMMA_F16(a, b1, acc1);
  }
  int rbase = tm * 16 + (lane >> 4) * 8;
  int col0 = h * CH + (lane & 15);
#pragma unroll
  for (int j = 0; j < 8; ++j) {
    msg16[(rbase + j) * DIM + col0]      = (_Float16)acc0[j];
    msg16[(rbase + j) * DIM + col0 + 16] = (_Float16)acc1[j];
  }
}

// ---------- Proj GEMM + bias + residual v ----------
__global__ __launch_bounds__(32) void proj_kernel(const _Float16* __restrict__ msg16,
                                                  const _Float16* __restrict__ wT,
                                                  const float* __restrict__ v32,
                                                  const float* __restrict__ bproj,
                                                  float* __restrict__ mout) {
  int tm = blockIdx.x, tn = blockIdx.y;
  int lane = threadIdx.x;
  int rowA = tm * 16 + (lane & 15);
  int colB = tn * 16 + (lane & 15);
  const _Float16* ap = msg16 + rowA * DIM + (lane >> 4) * 8;
  const _Float16* bp = wT + colB * DIM + (lane >> 4) * 8;
  v8f acc = {};
#pragma unroll
  for (int kb = 0; kb < DIM; kb += 32) {
    v16h a = ld_pair(ap + kb, ap + kb + 16);
    v16h b = ld_pair(bp + kb, bp + kb + 16);
    acc = WMMA_F16(a, b, acc);
  }
  int rbase = tm * 16 + (lane >> 4) * 8;
  float bb = bproj[colB];
#pragma unroll
  for (int j = 0; j < 8; ++j) {
    int row = rbase + j;
    mout[row * DIM + colB] = v32[row * DIM + colB] + acc[j] + bb;
  }
}

// ---------- Top-16 of A_mean rows (matches lax.top_k tie-break: lower index wins) ----------
__global__ __launch_bounds__(256) void topk_kernel(const float* __restrict__ Amean,
                                                   int* __restrict__ outIdx) {
  __shared__ float rowv[N_TOK];
  __shared__ float bv[256];
  __shared__ int   bi[256];
  int row = blockIdx.x, tid = threadIdx.x;
  const float* arow = Amean + (size_t)row * N_TOK;
#pragma unroll
  for (int i = 0; i < 8; ++i) rowv[i * 256 + tid] = arow[i * 256 + tid];
  __syncthreads();
  for (int sel = 0; sel < TOPK; ++sel) {
    float best = -3.4e38f; int besti = 0x7fffffff;
#pragma unroll
    for (int i = 0; i < 8; ++i) {
      int idx = i * 256 + tid;
      float val = rowv[idx];
      if (val > best || (val == best && idx < besti)) { best = val; besti = idx; }
    }
    bv[tid] = best; bi[tid] = besti; __syncthreads();
    for (int s = 128; s; s >>= 1) {
      if (tid < s) {
        if (bv[tid + s] > bv[tid] || (bv[tid + s] == bv[tid] && bi[tid + s] < bi[tid])) {
          bv[tid] = bv[tid + s]; bi[tid] = bi[tid + s];
        }
      }
      __syncthreads();
    }
    if (tid == 0) { outIdx[row * TOPK + sel] = bi[0]; rowv[bi[0]] = -3.4e38f; }
    __syncthreads();
  }
}

extern "C" void kernel_launch(void* const* d_in, const int* in_sizes, int n_in,
                              void* d_out, int out_size, void* d_ws, size_t ws_size,
                              hipStream_t stream) {
  (void)in_sizes; (void)n_in; (void)out_size; (void)ws_size;
  const float* point  = (const float*)d_in[0];
  const float* w_qkv  = (const float*)d_in[1];
  const float* w_proj = (const float*)d_in[2];
  const float* b_proj = (const float*)d_in[3];
  const float* n1w    = (const float*)d_in[4];
  const float* n1b    = (const float*)d_in[5];
  const float* n2w    = (const float*)d_in[6];
  const float* n2b    = (const float*)d_in[7];

  char* base = (char*)d_ws;
  size_t off = 0;
  auto alloc = [&](size_t bytes) -> char* {
    char* p = base + off;
    off = (off + bytes + 255) & ~(size_t)255;
    return p;
  };
  _Float16* x16    = (_Float16*)alloc((size_t)N_TOK * IN_DIM * 2);
  _Float16* wqkvT  = (_Float16*)alloc((size_t)QKV_DIM * IN_DIM * 2);
  _Float16* wprojT = (_Float16*)alloc((size_t)DIM * DIM * 2);
  _Float16* q16    = (_Float16*)alloc((size_t)N_TOK * DIM * 2);
  _Float16* k16    = (_Float16*)alloc((size_t)N_TOK * DIM * 2);
  _Float16* vT16   = (_Float16*)alloc((size_t)DIM * N_TOK * 2);
  float*    v32    = (float*)alloc((size_t)N_TOK * DIM * 4);
  float*    S      = (float*)alloc((size_t)N_TOK * N_TOK * 4);
  _Float16* P16    = (_Float16*)alloc((size_t)N_TOK * N_TOK * 2);
  float*    Amean  = (float*)alloc((size_t)N_TOK * N_TOK * 4);
  _Float16* msg16  = (_Float16*)alloc((size_t)N_TOK * DIM * 2);
  float*    mout   = (float*)alloc((size_t)N_TOK * DIM * 4);

  float* out   = (float*)d_out;
  int*   oIdx  = (int*)(out + (size_t)N_TOK * DIM);

  ln1_kernel<<<N_TOK, 256, 0, stream>>>(point, n1w, n1b, x16);
  convw_kernel<<<(QKV_DIM * IN_DIM + 255) / 256, 256, 0, stream>>>(w_qkv, w_proj, wqkvT, wprojT);
  qkv_gemm_kernel<<<dim3(N_TOK / 16, QKV_DIM / 16), 32, 0, stream>>>(x16, wqkvT, q16, k16, vT16, v32);
  for (int h = 0; h < NHEADS; ++h) {
    qk_kernel<<<dim3(N_TOK / 16, N_TOK / 64), 32, 0, stream>>>(q16, k16, S, h);
    softmax_kernel<<<N_TOK, 256, 0, stream>>>(S, P16, Amean, h == 0 ? 1 : 0);
    av_kernel<<<N_TOK / 16, 32, 0, stream>>>(P16, vT16, msg16, h);
  }
  proj_kernel<<<dim3(N_TOK / 16, DIM / 16), 32, 0, stream>>>(msg16, wprojT, v32, b_proj, mout);
  ln2_kernel<<<N_TOK, 256, 0, stream>>>(mout, n2w, n2b, out);
  topk_kernel<<<N_TOK, 256, 0, stream>>>(Amean, oIdx);
}
